// AttnDecoderRNN_34256659153503
// MI455X (gfx1250) — matrix operand order repaired and checked by reading, not compiled, not run on hardware
//
#include <hip/hip_runtime.h>
#include <hip/hip_bf16.h>
#include <math.h>

// ---- problem constants (match reference) ----
#define BB 64
#define SS 64
#define TT 64
#define HH 512
#define VV 32000

typedef __bf16 bf16x8  __attribute__((ext_vector_type(8)));
typedef __bf16 bf16x16 __attribute__((ext_vector_type(16)));
typedef float  f32x8   __attribute__((ext_vector_type(8)));

union FragB16 { bf16x16 v; bf16x8 h[2]; };

// ---------------------------------------------------------------------------
// bf16 WMMA GEMM:  C[M,N] = A[M,K] * W[N,K]^T + bias[N]
// A row-major [M,K] bf16, W row-major [N,K] bf16, C f32 with row stride ldc.
// Tiling chosen so the LARGE matrix (W) is streamed from L2 exactly once:
//   - each wave owns ONE 16-column tile of W and accumulates FOUR 16-row
//     M tiles (a 64-row slab), sharing the B fragment across 4 WMMAs.
//   - A (the small activation matrix) is re-read per wave but is WGP-cache
//     resident (64 KB for the decode GEMMs).
// Grid: x = N/128 (8 waves/block, 16 cols each), y = M/64. Block: 256 thr.
// K must be a multiple of 64. Double-buffered over K in steps of 64.
// ---------------------------------------------------------------------------
__global__ void gemm_bf16_wmma(const __bf16* __restrict__ A,
                               const __bf16* __restrict__ W,
                               const float*  __restrict__ bias,
                               float* __restrict__ C,
                               long ldc, int K) {
  const int lane = threadIdx.x & 31;
  const int wave = threadIdx.x >> 5;
  const long n0  = ((long)blockIdx.x * 8 + wave) * 16;   // this wave's N tile
  const int  m0  = blockIdx.y * 64;                      // 64-row M slab

  f32x8 acc[4] = {};

  // 16-bit A-matrix fragment layout (16x32): row = lane&15,
  //   chunks at k + (lane<16?0:8) and k+16+(lane<16?0:8)
  const int rowA  = lane & 15;
  const int kaoff = (lane & 16) ? 8 : 0;
  // 16-bit B-matrix fragment layout (32x16): col = lane&15,
  //   chunks at k + (lane<16?0:8) and k+16+(lane<16?0:8)
  const int colB  = lane & 15;
  const int kboff = (lane & 16) ? 8 : 0;

  const __bf16* pa = A + (size_t)(m0 + rowA) * K + kaoff;
  const __bf16* pb = W + (size_t)(n0 + colB) * K + kboff;
  const size_t aRow16 = (size_t)16 * K;   // 16 rows of A

  FragB16 aX[4], aY[4], bX, bY;

  // prologue: chunk 0 -> X buffers
  bX.h[0] = *(const bf16x8*)(pb);
  bX.h[1] = *(const bf16x8*)(pb + 16);
#pragma unroll
  for (int i = 0; i < 4; ++i) {
    aX[i].h[0] = *(const bf16x8*)(pa + i * aRow16);
    aX[i].h[1] = *(const bf16x8*)(pa + i * aRow16 + 16);
  }

  for (int kk = 0; kk < K; kk += 64) {
    // prefetch chunk kk+32 -> Y while X is consumed
    bY.h[0] = *(const bf16x8*)(pb + kk + 32);
    bY.h[1] = *(const bf16x8*)(pb + kk + 48);
#pragma unroll
    for (int i = 0; i < 4; ++i) {
      aY[i].h[0] = *(const bf16x8*)(pa + i * aRow16 + kk + 32);
      aY[i].h[1] = *(const bf16x8*)(pa + i * aRow16 + kk + 48);
    }
#pragma unroll
    for (int i = 0; i < 4; ++i)
      acc[i] = __builtin_amdgcn_wmma_f32_16x16x32_bf16(
          false, aX[i].v, false, bX.v, (short)0, acc[i], false, false);

    // prefetch chunk kk+64 -> X while Y is consumed
    if (kk + 64 < K) {
      bX.h[0] = *(const bf16x8*)(pb + kk + 64);
      bX.h[1] = *(const bf16x8*)(pb + kk + 80);
#pragma unroll
      for (int i = 0; i < 4; ++i) {
        aX[i].h[0] = *(const bf16x8*)(pa + i * aRow16 + kk + 64);
        aX[i].h[1] = *(const bf16x8*)(pa + i * aRow16 + kk + 80);
      }
    }
#pragma unroll
    for (int i = 0; i < 4; ++i)
      acc[i] = __builtin_amdgcn_wmma_f32_16x16x32_bf16(
          false, aY[i].v, false, bY.v, (short)0, acc[i], false, false);
  }

  // C/D layout: col = lane&15; VGPR r -> row (lane<16 ? r : r+8)
  const int  colC = lane & 15;
  const int  rsub = (lane & 16) ? 8 : 0;
  const long ncol = n0 + colC;
  const float bval = bias[ncol];
#pragma unroll
  for (int i = 0; i < 4; ++i) {
    const long rbase = m0 + i * 16 + rsub;
#pragma unroll
    for (int r = 0; r < 8; ++r)
      C[(size_t)(rbase + r) * ldc + ncol] = acc[i][r] + bval;
  }
}

// ---- f32 -> bf16 bulk convert ----
__global__ void cvt_f32_bf16(const float* __restrict__ src,
                             __bf16* __restrict__ dst, int n) {
  int i = blockIdx.x * blockDim.x + threadIdx.x;
  if (i < n) dst[i] = (__bf16)src[i];
}

// ---- teacher-forced embedding gather (SOS=0 then target[:, :-1]) ----
__global__ void gather_emb(const int* __restrict__ tgt,
                           const float* __restrict__ emb,
                           __bf16* __restrict__ dst) {
  const int bt = blockIdx.x;          // b*T + t
  const int b  = bt / TT;
  const int t  = bt % TT;
  const int id = (t == 0) ? 0 : tgt[b * TT + (t - 1)];
  const float* src = emb + (size_t)id * HH;
  __bf16* d = dst + (size_t)bt * HH;
  for (int i = threadIdx.x; i < HH; i += blockDim.x) d[i] = (__bf16)src[i];
}

// ---- h0 init: f32 copy + bf16 shadow ----
__global__ void init_h(const float* __restrict__ eh,
                       float* __restrict__ h, __bf16* __restrict__ hb) {
  int i = blockIdx.x * blockDim.x + threadIdx.x;   // B*H
  float v = eh[i];
  h[i] = v;
  hb[i] = (__bf16)v;
}

// ---------------------------------------------------------------------------
// Bahdanau attention for one step. One block per batch element, 256 threads.
// scores[s] = Va . tanh(q[b] + Uk[b,s]) + bv ; softmax ; ctx = w . enc[b]
// Writes attention weights to d_out and x = [emb_t | ctx] in bf16.
// ---------------------------------------------------------------------------
__global__ void attn_step(const float* __restrict__ q,
                          const float* __restrict__ Uk,
                          const float* __restrict__ enc,
                          const float* __restrict__ Va,
                          const float* __restrict__ bv,
                          const __bf16* __restrict__ emb_seq,
                          __bf16* __restrict__ x,
                          float* __restrict__ attn_out,
                          int t) {
  __shared__ float sc[SS];
  const int b    = blockIdx.x;
  const int tid  = threadIdx.x;
  const int lane = tid & 31;
  const int w    = tid >> 5;          // 8 waves

  const float* qb = q + (size_t)b * HH;
  // 8 scores per wave, 32-lane reduction over H
  for (int s = w; s < SS; s += 8) {
    const float* uk = Uk + ((size_t)b * SS + s) * HH;
    float part = 0.f;
    for (int hh = lane; hh < HH; hh += 32)
      part += tanhf(qb[hh] + uk[hh]) * Va[hh];
    for (int off = 16; off > 0; off >>= 1) part += __shfl_down(part, off);
    if (lane == 0) sc[s] = part + bv[0];
  }
  __syncthreads();

  // softmax over 64 scores in wave 0
  if (w == 0) {
    float v0 = sc[lane], v1 = sc[lane + 32];
    float m = fmaxf(v0, v1);
    for (int off = 16; off > 0; off >>= 1) m = fmaxf(m, __shfl_xor(m, off));
    float e0 = expf(v0 - m), e1 = expf(v1 - m);
    float sum = e0 + e1;
    for (int off = 16; off > 0; off >>= 1) sum += __shfl_xor(sum, off);
    float inv = 1.f / sum;
    sc[lane] = e0 * inv;
    sc[lane + 32] = e1 * inv;
  }
  __syncthreads();

  if (tid < SS) attn_out[((size_t)b * TT + t) * SS + tid] = sc[tid];

  // ctx + build x = [emb_t | ctx]
  const __bf16* et = emb_seq + ((size_t)b * TT + t) * HH;
  for (int hh = tid; hh < HH; hh += blockDim.x) {
    float c = 0.f;
    const float* eb = enc + (size_t)b * SS * HH + hh;
#pragma unroll 4
    for (int s = 0; s < SS; ++s) c += sc[s] * eb[(size_t)s * HH];
    x[(size_t)b * (2 * HH) + hh]      = et[hh];
    x[(size_t)b * (2 * HH) + HH + hh] = (__bf16)c;
  }
}

// ---- GRU gate elementwise (PyTorch r,z,n layout) ----
__global__ void gru_gate(const float* __restrict__ gi,
                         const float* __restrict__ gh,
                         float* __restrict__ h, __bf16* __restrict__ hb) {
  int i  = blockIdx.x * blockDim.x + threadIdx.x;   // B*H
  int b  = i >> 9;
  int hh = i & (HH - 1);
  const float* gib = gi + (size_t)b * (3 * HH);
  const float* ghb = gh + (size_t)b * (3 * HH);
  float r = 1.f / (1.f + expf(-(gib[hh] + ghb[hh])));
  float z = 1.f / (1.f + expf(-(gib[HH + hh] + ghb[HH + hh])));
  float n = tanhf(gib[2 * HH + hh] + r * ghb[2 * HH + hh]);
  float hp = h[i];
  float hn = (1.f - z) * n + z * hp;
  h[i] = hn;
  hb[i] = (__bf16)hn;
}

__global__ void copy_f32(const float* __restrict__ src,
                         float* __restrict__ dst, int n) {
  int i = blockIdx.x * blockDim.x + threadIdx.x;
  if (i < n) dst[i] = src[i];
}

// ---- in-place log_softmax over rows of length V ----
__global__ void log_softmax_rows(float* __restrict__ out) {
  __shared__ float red[256];
  float* x = out + (size_t)blockIdx.x * VV;
  const int tid = threadIdx.x;

  float m = -INFINITY;
  for (int i = tid; i < VV; i += 256) m = fmaxf(m, x[i]);
  red[tid] = m; __syncthreads();
  for (int s = 128; s > 0; s >>= 1) {
    if (tid < s) red[tid] = fmaxf(red[tid], red[tid + s]);
    __syncthreads();
  }
  m = red[0]; __syncthreads();

  float sum = 0.f;
  for (int i = tid; i < VV; i += 256) sum += expf(x[i] - m);
  red[tid] = sum; __syncthreads();
  for (int s = 128; s > 0; s >>= 1) {
    if (tid < s) red[tid] += red[tid + s];
    __syncthreads();
  }
  const float lse = m + logf(red[0]);

  for (int i = tid; i < VV; i += 256) x[i] = x[i] - lse;
}

// ---------------------------------------------------------------------------
extern "C" void kernel_launch(void* const* d_in, const int* in_sizes, int n_in,
                              void* d_out, int out_size, void* d_ws, size_t ws_size,
                              hipStream_t stream) {
  (void)in_sizes; (void)n_in; (void)out_size; (void)ws_size;

  const float* enc  = (const float*)d_in[0];   // [B,S,H]
  const float* eh   = (const float*)d_in[1];   // [1,B,H]
  const int*   tgt  = (const int*)  d_in[2];   // [B,T]
  const float* emb  = (const float*)d_in[3];   // [V,H]
  const float* Wa   = (const float*)d_in[4];   // [H,H]
  const float* ba   = (const float*)d_in[5];   // [H]
  const float* Ua   = (const float*)d_in[6];   // [H,H]
  const float* bu   = (const float*)d_in[7];   // [H]
  const float* Va   = (const float*)d_in[8];   // [1,H]
  const float* bv   = (const float*)d_in[9];   // [1]
  const float* Wih  = (const float*)d_in[10];  // [3H,2H]
  const float* Whh  = (const float*)d_in[11];  // [3H,H]
  const float* bih  = (const float*)d_in[12];  // [3H]
  const float* bhh  = (const float*)d_in[13];  // [3H]
  const float* Wout = (const float*)d_in[14];  // [V,H]
  const float* bout = (const float*)d_in[15];  // [V]

  float* out = (float*)d_out;
  const size_t OUT_LOGP = 0;
  const size_t OUT_HT   = (size_t)BB * TT * VV;             // 131072000
  const size_t OUT_ATTN = OUT_HT + (size_t)BB * HH;         // +32768

  // ---- workspace carve (all sizes multiples of 256B) ----
  char* ws = (char*)d_ws;
  size_t off = 0;
  auto carve = [&](size_t bytes) { void* p = ws + off; off += bytes; return p; };

  __bf16* wa_b   = (__bf16*)carve((size_t)HH * HH * 2);
  __bf16* ua_b   = (__bf16*)carve((size_t)HH * HH * 2);
  __bf16* wih_b  = (__bf16*)carve((size_t)3 * HH * 2 * HH * 2);
  __bf16* whh_b  = (__bf16*)carve((size_t)3 * HH * HH * 2);
  __bf16* wout_b = (__bf16*)carve((size_t)VV * HH * 2);
  __bf16* enc_b  = (__bf16*)carve((size_t)BB * SS * HH * 2);
  __bf16* embs_b = (__bf16*)carve((size_t)BB * TT * HH * 2);
  float*  Uk     = (float*) carve((size_t)BB * SS * HH * 4);
  float*  hbuf   = (float*) carve((size_t)BB * HH * 4);
  __bf16* h_b    = (__bf16*)carve((size_t)BB * HH * 2);
  float*  qbuf   = (float*) carve((size_t)BB * HH * 4);
  __bf16* x_b    = (__bf16*)carve((size_t)BB * 2 * HH * 2);
  float*  gi     = (float*) carve((size_t)BB * 3 * HH * 4);
  float*  gh     = (float*) carve((size_t)BB * 3 * HH * 4);

  // ---- one-time setup ----
  const int thr = 256;
  cvt_f32_bf16<<<(HH * HH + thr - 1) / thr, thr, 0, stream>>>(Wa, wa_b, HH * HH);
  cvt_f32_bf16<<<(HH * HH + thr - 1) / thr, thr, 0, stream>>>(Ua, ua_b, HH * HH);
  cvt_f32_bf16<<<(3 * HH * 2 * HH + thr - 1) / thr, thr, 0, stream>>>(Wih, wih_b, 3 * HH * 2 * HH);
  cvt_f32_bf16<<<(3 * HH * HH + thr - 1) / thr, thr, 0, stream>>>(Whh, whh_b, 3 * HH * HH);
  cvt_f32_bf16<<<(VV * HH + thr - 1) / thr, thr, 0, stream>>>(Wout, wout_b, VV * HH);
  cvt_f32_bf16<<<(BB * SS * HH + thr - 1) / thr, thr, 0, stream>>>(enc, enc_b, BB * SS * HH);
  gather_emb<<<BB * TT, 256, 0, stream>>>(tgt, emb, embs_b);
  init_h<<<(BB * HH) / 256, 256, 0, stream>>>(eh, hbuf, h_b);

  // Uk = enc @ Ua^T + bu : M=4096, N=512, K=512
  gemm_bf16_wmma<<<dim3(HH / 128, (BB * SS) / 64), 256, 0, stream>>>(
      enc_b, ua_b, bu, Uk, HH, HH);

  // ---- sequential decode ----
  for (int t = 0; t < TT; ++t) {
    // q = h @ Wa^T + ba : M=64, N=512, K=512
    gemm_bf16_wmma<<<dim3(HH / 128, 1), 256, 0, stream>>>(
        h_b, wa_b, ba, qbuf, HH, HH);
    // attention + build x
    attn_step<<<BB, 256, 0, stream>>>(qbuf, Uk, enc, Va, bv, embs_b, x_b,
                                      out + OUT_ATTN, t);
    // gi = x @ W_ih^T + b_ih : M=64, N=1536, K=1024
    gemm_bf16_wmma<<<dim3((3 * HH) / 128, 1), 256, 0, stream>>>(
        x_b, wih_b, bih, gi, 3 * HH, 2 * HH);
    // gh = h @ W_hh^T + b_hh : M=64, N=1536, K=512
    gemm_bf16_wmma<<<dim3((3 * HH) / 128, 1), 256, 0, stream>>>(
        h_b, whh_b, bhh, gh, 3 * HH, HH);
    // gates -> h_new (f32 + bf16 shadow)
    gru_gate<<<(BB * HH) / 256, 256, 0, stream>>>(gi, gh, hbuf, h_b);
    // logits = h_new @ W_out^T + b_out, written into d_out[b, t, :] (ldc = T*V)
    gemm_bf16_wmma<<<dim3(VV / 128, 1), 256, 0, stream>>>(
        h_b, wout_b, bout, out + OUT_LOGP + (size_t)t * VV, (long)TT * VV, HH);
  }

  // hT -> d_out
  copy_f32<<<(BB * HH) / 256, 256, 0, stream>>>(hbuf, out + OUT_HT, BB * HH);
  // in-place log_softmax over [B*T, V]
  log_softmax_rows<<<BB * TT, 256, 0, stream>>>(out + OUT_LOGP);
}
